// CombineLossV1_72722386256449
// MI455X (gfx1250) — matrix operand order repaired and checked by reading, not compiled, not run on hardware
//
#include <hip/hip_runtime.h>
#include <hip/hip_bf16.h>

// Problem constants (from reference)
#define N_ROWS 1024
#define DDIM   2048
#define NATTR  40
#define NCOL   80          // 2*NATTR
#define KSPLIT 8           // K chunks of 256
#define KCHUNK (DDIM / KSPLIT)

typedef __attribute__((ext_vector_type(16))) _Float16 v16h;
typedef __attribute__((ext_vector_type(8)))  float    v8f;
typedef __attribute__((ext_vector_type(4)))  float    v4f;

// ---------------- Kernel 1: fold maskweight into clsweight, transpose, f16 ----
// Wt[j][d] = (f16)(maskweight[j/2][d] * clsweight[d][j]),  j in [0,80), d in [0,2048)
__global__ __launch_bounds__(256) void prep_w_kernel(
    const float* __restrict__ maskweight,   // (40, 2048)
    const float* __restrict__ clsweight,    // (2048, 80)
    _Float16* __restrict__ Wt)              // (80, 2048) f16
{
    int idx = blockIdx.x * 256 + threadIdx.x;
    if (idx >= NCOL * DDIM) return;
    int j = idx >> 11;          // / 2048
    int d = idx & (DDIM - 1);   // % 2048
    float v = maskweight[(j >> 1) * DDIM + d] * clsweight[(size_t)d * NCOL + j];
    Wt[(size_t)j * DDIM + d] = (_Float16)v;
}

// ---------------- Kernel 2: split-K WMMA GEMM, partial logits to ws ----------
// grid: 64 blocks x 256 threads = 512 waves; wave w: mtile = w % 64, kchunk = w / 64
__global__ __launch_bounds__(256) void gemm_wmma_kernel(
    const float* __restrict__ X,            // (1024, 2048) f32
    const _Float16* __restrict__ Wt,        // (80, 2048) f16
    float* __restrict__ part)               // (KSPLIT, 1024, 80) f32 partial logits
{
    const int lane  = threadIdx.x & 31;
    const int wavei = threadIdx.x >> 5;
    const int w     = blockIdx.x * 8 + wavei;
    const int mtile  = w & 63;    // 64 row tiles of 16
    const int kchunk = w >> 6;    // 8 K chunks
    const int row = lane & 15;
    const bool hi = lane >= 16;
    const int m = mtile * 16 + row;           // A row for this lane
    const int kbegin = kchunk * KCHUNK;

    v8f acc[5] = {};                          // 5 column tiles (N = 80)

    const float* xrow = X + (size_t)m * DDIM;
    const int a_off = hi ? 16 : 0;            // A layout: hi lanes K=16..31 (matches B)
    const int b_off = hi ? 16 : 0;            // B layout used here: hi lanes K=16..31 (matches A)

    for (int k0 = kbegin; k0 < kbegin + KCHUNK; k0 += 32) {
        // ---- A fragment: 16 halves/lane, K = base..base+15
        const int ab = k0 + a_off;
        v4f x0 = *(const v4f*)(xrow + ab);
        v4f x1 = *(const v4f*)(xrow + ab + 4);
        v4f x2 = *(const v4f*)(xrow + ab + 8);
        v4f x3 = *(const v4f*)(xrow + ab + 12);
        v16h a;
        #pragma unroll
        for (int i = 0; i < 4; ++i) {
            a[i]      = (_Float16)x0[i];
            a[4 + i]  = (_Float16)x1[i];
            a[8 + i]  = (_Float16)x2[i];
            a[12 + i] = (_Float16)x3[i];
        }
        const int kb = k0 + b_off;
        #pragma unroll
        for (int t = 0; t < 5; ++t) {
            // B fragment: column j = t*16 + row, 16 contiguous K halves (32B aligned)
            const v16h b = *(const v16h*)(Wt + (size_t)(t * 16 + row) * DDIM + kb);
            acc[t] = __builtin_amdgcn_wmma_f32_16x16x32_f16(
                false, a, false, b, (short)0, acc[t], false, false);
        }
    }

    // ---- store partial logits; C/D layout: VGPR r -> M = r (+8 for hi lanes)
    float* pbase = part + (size_t)kchunk * N_ROWS * NCOL;
    const int nbase = mtile * 16 + (hi ? 8 : 0);
    const int col   = row;
    #pragma unroll
    for (int t = 0; t < 5; ++t) {
        #pragma unroll
        for (int r = 0; r < 8; ++r) {
            pbase[(size_t)(nbase + r) * NCOL + t * 16 + col] = acc[t][r];
        }
    }
}

// ---------------- Kernel 3: per-row cross-entropy over 40 attribute pairs ----
__global__ __launch_bounds__(256) void loss_rows_kernel(
    const float* __restrict__ part,         // (KSPLIT, 1024, 80)
    const int* __restrict__ target,         // (1024, 40)
    float* __restrict__ rowloss)            // (1024,)
{
    int n = blockIdx.x * 256 + threadIdx.x;
    if (n >= N_ROWS) return;
    float s = 0.0f;
    for (int a = 0; a < NATTR; ++a) {
        float l0 = 0.0f, l1 = 0.0f;
        #pragma unroll
        for (int k = 0; k < KSPLIT; ++k) {   // fixed-order K reduction: deterministic
            const float* p = part + ((size_t)k * N_ROWS + n) * NCOL + 2 * a;
            l0 += p[0];
            l1 += p[1];
        }
        float mx  = fmaxf(l0, l1);
        float lse = mx + __logf(__expf(l0 - mx) + __expf(l1 - mx));
        int   t   = target[(size_t)n * NATTR + a];
        s += lse - (t ? l1 : l0);
    }
    rowloss[n] = s;
}

// ---------------- Kernel 4: final deterministic reduction ------------------
__global__ __launch_bounds__(256) void reduce_kernel(
    const float* __restrict__ rowloss, float* __restrict__ out)
{
    __shared__ float sm[256];
    float s = 0.0f;
    for (int i = threadIdx.x; i < N_ROWS; i += 256) s += rowloss[i];
    sm[threadIdx.x] = s;
    __syncthreads();
    for (int off = 128; off > 0; off >>= 1) {
        if (threadIdx.x < (unsigned)off) sm[threadIdx.x] += sm[threadIdx.x + off];
        __syncthreads();
    }
    if (threadIdx.x == 0) out[0] = sm[0] * (1.0f / (float)N_ROWS);
}

extern "C" void kernel_launch(void* const* d_in, const int* in_sizes, int n_in,
                              void* d_out, int out_size, void* d_ws, size_t ws_size,
                              hipStream_t stream) {
    const float* globalfea  = (const float*)d_in[0];   // (1024, 2048)
    const float* maskweight = (const float*)d_in[1];   // (40, 2048)
    const float* clsweight  = (const float*)d_in[2];   // (2048, 80)
    const int*   target     = (const int*)d_in[3];     // (1024, 40)
    float* out = (float*)d_out;

    // Workspace layout
    const size_t WT_BYTES   = (size_t)NCOL * DDIM * sizeof(_Float16);        // 320 KB
    const size_t PART_BYTES = (size_t)KSPLIT * N_ROWS * NCOL * sizeof(float); // 2.5 MB
    _Float16* Wt      = (_Float16*)d_ws;
    float*    part    = (float*)((char*)d_ws + WT_BYTES);
    float*    rowloss = (float*)((char*)d_ws + WT_BYTES + PART_BYTES);

    // 1) fold + transpose weights to f16
    prep_w_kernel<<<(NCOL * DDIM + 255) / 256, 256, 0, stream>>>(maskweight, clsweight, Wt);
    // 2) split-K WMMA GEMM: 64 blocks x 8 waves = 512 waves
    gemm_wmma_kernel<<<64, 256, 0, stream>>>(globalfea, Wt, part);
    // 3) per-row loss
    loss_rows_kernel<<<(N_ROWS + 255) / 256, 256, 0, stream>>>(part, target, rowloss);
    // 4) scalar reduce
    reduce_kernel<<<1, 256, 0, stream>>>(rowloss, out);
}
